// AutoCorrProj_43155831390613
// MI455X (gfx1250) — matrix operand order repaired and checked by reading, not compiled, not run on hardware
//
#include <hip/hip_runtime.h>

#define B_   4
#define CIN  256
#define H_   96
#define W_   96
#define CC_  64
#define OC_  32
#define K1TOT (CIN*9)    // 2304 = c*9 + (u*3+v), matches extract_w flat layout
#define K2TOT (CC_*25)   // 1600 = c*25 + (u*5+v), matches reg_w flat layout

// workspace layout (bytes)
#define WS_F_OFF    0
#define WS_EWB_OFF  (B_*CC_*H_*W_*4)                 // 9,437,184
#define WS_RWB_OFF  (WS_EWB_OFF + CC_*K1TOT*2)      // + 294,912

#define AS1 __attribute__((address_space(1)))
#define AS3 __attribute__((address_space(3)))

#if defined(__has_builtin)
#  if __has_builtin(__builtin_amdgcn_global_load_async_to_lds_b128)
#    define ASYNC_LDS 1
#  endif
#endif

typedef __attribute__((ext_vector_type(16))) __bf16       v16bf;
typedef __attribute__((ext_vector_type(8)))  float        v8f;
typedef __attribute__((ext_vector_type(4)))  unsigned int v4u;
typedef __attribute__((ext_vector_type(4)))  int          v4i;

union FragCast { v4u u4[2]; v16bf b16; };

// f32 -> bf16, round to nearest even
__device__ __forceinline__ unsigned short f2bf(float f) {
    unsigned int u = __float_as_uint(f);
    u += 0x7FFFu + ((u >> 16) & 1u);
    return (unsigned short)(u >> 16);
}
__device__ __forceinline__ unsigned int f2bf2(float lo, float hi) {
    return (unsigned int)f2bf(lo) | ((unsigned int)f2bf(hi) << 16);
}

// Copy 16B global(bf16-packed) -> LDS; async DMA path when available.
__device__ __forceinline__ void cp16_g2l(unsigned short* dst, const unsigned short* src) {
#ifdef ASYNC_LDS
    __builtin_amdgcn_global_load_async_to_lds_b128(
        (AS1 v4i*)(unsigned short*)src, (AS3 v4i*)dst, 0, 0);
#else
    *(v4u*)dst = *(const v4u*)src;
#endif
}
__device__ __forceinline__ void cp_drain() {
#ifdef ASYNC_LDS
#  if __has_builtin(__builtin_amdgcn_s_wait_asynccnt)
    __builtin_amdgcn_s_wait_asynccnt(0);
#  else
    asm volatile("s_wait_asynccnt 0x0" ::: "memory");
#  endif
#endif
}

// A fragment (16x32 bf16): lane (row m, half g):
//   VGPR0-7 = k [16g .. 16g+15]
__device__ __forceinline__ v16bf load_frag_a(const unsigned short* base, int row, int g) {
    FragCast fc;
    fc.u4[0] = *(const v4u*)(base + row*32 + 16*g);
    fc.u4[1] = *(const v4u*)(base + row*32 + 16*g + 8);
    return fc.b16;
}
// B fragment (32x16 bf16): lane (col n, half g): VGPR0-7 = k [16g .. 16g+15]
__device__ __forceinline__ v16bf load_frag_b(const unsigned short* base, int col, int g) {
    FragCast fc;
    fc.u4[0] = *(const v4u*)(base + col*32 + 16*g);
    fc.u4[1] = *(const v4u*)(base + col*32 + 16*g + 8);
    return fc.b16;
}

// ---------------------------------------------------------------------------
// Kernel 0: one-time f32 -> bf16 pre-pack of both weight tensors.
// ---------------------------------------------------------------------------
#define EW_PAIRS (CC_*K1TOT/2)   // 73728
#define RW_PAIRS (OC_*K2TOT/2)   // 25600
__global__ __launch_bounds__(256) void pack_weights(
    const float* __restrict__ ew, const float* __restrict__ rw,
    unsigned short* __restrict__ ewb, unsigned short* __restrict__ rwb)
{
    int p = blockIdx.x*256 + threadIdx.x;
    if (p < EW_PAIRS) {
        *(unsigned int*)(ewb + 2*p) = f2bf2(ew[2*p], ew[2*p + 1]);
    } else if (p < EW_PAIRS + RW_PAIRS) {
        int q = p - EW_PAIRS;
        *(unsigned int*)(rwb + 2*q) = f2bf2(rw[2*q], rw[2*q + 1]);
    }
}

// ---------------------------------------------------------------------------
// Kernel 1: 3x3 conv (Cin=256 -> CC=64) as implicit GEMM, one image row per
// block. M=64 (4 wave-strips of 16), N=96 (6 tiles of 16), K=2304.
// Per-k address table hoists all div/mod + bounds math out of the K loop.
// ---------------------------------------------------------------------------
__global__ __launch_bounds__(128) void conv3x3_wmma(
    const float* __restrict__ x, const unsigned short* __restrict__ wb,
    const float* __restrict__ bias, float* __restrict__ f)
{
    __shared__ __align__(16) int            tab[K1TOT*2];  // {rowOff, nLo|nHi<<8}
    __shared__ __align__(16) unsigned short As[CC_*32];    // [m][k]
    __shared__ __align__(16) unsigned short Bs[W_*32];     // [n][k] (im2col tile)

    const int tid  = threadIdx.x;
    const int lane = tid & 31;
    const int wv   = tid >> 5;     // 0..3 -> 16-row m strip
    const int g    = lane >> 4;
    const int li   = lane & 15;

    const int bi = blockIdx.x;     // b*H + i
    const int b  = bi / H_;
    const int i  = bi - b*H_;

    // Fill k -> (global row offset, valid n range) table once.
    for (int k = tid; k < K1TOT; k += 128) {
        int c  = k / 9;
        int t9 = k - 9*c;
        int u  = t9 / 3;
        int v  = t9 - 3*u;
        int rr = i + u - 1;
        int rowOff = ((b*CIN + c)*H_ + rr)*W_ + (v - 1);
        int nLo = (1 - v) > 0 ? (1 - v) : 0;          // n + v - 1 >= 0
        int nHi = (97 - v) < W_ ? (97 - v) : W_;      // n + v - 1 < 96
        if (rr < 0 || rr >= H_) { nLo = W_; nHi = 0; }
        tab[2*k]     = rowOff;
        tab[2*k + 1] = nLo | (nHi << 8);
    }

    v8f acc[6] = {};
    float bm[8];
#pragma unroll
    for (int r = 0; r < 8; ++r) bm[r] = bias[wv*16 + r + 8*g];

    __syncthreads();

    for (int k0 = 0; k0 < K1TOT; k0 += 32) {
        // Stage A: raw 16B copies of pre-packed bf16 weight rows (async DMA
        // to LDS when the toolchain exposes it). 64 rows x 64B = 256 chunks.
#pragma unroll
        for (int e = tid; e < 256; e += 128) {
            int row = e >> 2, part = e & 3;
            const unsigned short* src = wb + row*K1TOT + k0 + part*8;
            cp16_g2l(As + row*32 + part*8, src);
            if (k0 + 32 < K1TOT)
                __builtin_prefetch(src + 32, 0, 0);   // next chunk's A rows
        }
        // Stage B: im2col via table; two k's per iteration, one b32 LDS store.
        for (int e = tid; e < 16*W_; e += 128) {      // 1536 (kl-pair, n)
            int klp = e / W_;
            int n   = e - klp*W_;
            v4i t = *(const v4i*)(tab + 2*(k0 + 2*klp));
            float v0 = 0.0f, v1 = 0.0f;
            int rg0 = t[1], rg1 = t[3];
            if (n >= (rg0 & 0xFF) && n < (rg0 >> 8)) v0 = x[t[0] + n];
            if (n >= (rg1 & 0xFF) && n < (rg1 >> 8)) v1 = x[t[2] + n];
            *(unsigned int*)(Bs + n*32 + 2*klp) = f2bf2(v0, v1);
        }
        cp_drain();
        __syncthreads();

        v16bf afrag = load_frag_a(As, wv*16 + li, g);
#pragma unroll
        for (int nt = 0; nt < 6; ++nt) {
            v16bf bfrag = load_frag_b(Bs, nt*16 + li, g);
            acc[nt] = __builtin_amdgcn_wmma_f32_16x16x32_bf16(
                false, afrag, false, bfrag, (short)0, acc[nt], false, false);
        }
        __syncthreads();
    }

    // Epilogue: C/D layout -> lane holds (m = strip + r + 8g, n = ntile + li)
#pragma unroll
    for (int nt = 0; nt < 6; ++nt) {
#pragma unroll
        for (int r = 0; r < 8; ++r) {
            int m = wv*16 + r + 8*g;
            int n = nt*16 + li;
            f[((b*CC_ + m)*H_ + i)*W_ + n] = acc[nt][r] + bm[r];
        }
    }
}

// ---------------------------------------------------------------------------
// Kernel 2: autocorr projection as GEMM. y = W2 (32x1600) @ [f_shift * f_center]
// (1600 x 96). One image row per block; M=32 (2 strips), N=96 (2 halves x 3
// tiles), 4 waves. Per-k table: {centerOff, shiftOff, n-range}.
// ---------------------------------------------------------------------------
__global__ __launch_bounds__(128) void corrproj_wmma(
    const float* __restrict__ f, const unsigned short* __restrict__ wb,
    const float* __restrict__ rb, float* __restrict__ y)
{
    __shared__ __align__(16) int            tab[K2TOT*4];  // {cOff, sOff, range, 0}
    __shared__ __align__(16) unsigned short As[OC_*32];    // [m][k]
    __shared__ __align__(16) unsigned short Bs[W_*32];     // [n][k]

    const int tid   = threadIdx.x;
    const int lane  = tid & 31;
    const int wv    = tid >> 5;
    const int strip = wv & 1;      // m strip (0..1)
    const int nh    = wv >> 1;     // n half  (0..1), 48 columns each
    const int g     = lane >> 4;
    const int li    = lane & 15;

    const int bi = blockIdx.x;
    const int b  = bi / H_;
    const int i  = bi - b*H_;

    const float* fbase = f + (size_t)b * CC_ * H_ * W_;

    for (int k = tid; k < K2TOT; k += 128) {
        int c  = k / 25;
        int uv = k - 25*c;
        int u  = uv / 5;
        int v  = uv - 5*u;
        int rr = i + u - 2;
        int cOff = (c*H_ + i)*W_;
        int sOff = (c*H_ + rr)*W_ + (v - 2);
        int nLo = (2 - v) > 0 ? (2 - v) : 0;          // n + v - 2 >= 0
        int nHi = (98 - v) < W_ ? (98 - v) : W_;      // n + v - 2 < 96
        if (rr < 0 || rr >= H_) { nLo = W_; nHi = 0; }
        tab[4*k]     = cOff;
        tab[4*k + 1] = sOff;
        tab[4*k + 2] = nLo | (nHi << 8);
        tab[4*k + 3] = 0;
    }

    v8f acc[3] = {};
    float bm[8];
#pragma unroll
    for (int r = 0; r < 8; ++r) bm[r] = rb[strip*16 + r + 8*g];

    __syncthreads();

    for (int k0 = 0; k0 < K2TOT; k0 += 32) {
        // Stage A: raw 16B copies of pre-packed bf16 reg_w rows (128 chunks).
#pragma unroll
        for (int e = tid; e < 128; e += 128) {
            int o = e >> 2, part = e & 3;
            const unsigned short* src = wb + o*K2TOT + k0 + part*8;
            cp16_g2l(As + o*32 + part*8, src);
            if (k0 + 32 < K2TOT)
                __builtin_prefetch(src + 32, 0, 0);
        }
        // Stage B: correlation products via table; two k's per iteration.
        for (int e = tid; e < 16*W_; e += 128) {      // 1536 (kl-pair, n)
            int klp = e / W_;
            int n   = e - klp*W_;
            const int* tp = tab + 4*(k0 + 2*klp);
            v4i t0 = *(const v4i*)(tp);
            v4i t1 = *(const v4i*)(tp + 4);
            float p0 = 0.0f, p1 = 0.0f;
            if (n >= (t0[2] & 0xFF) && n < (t0[2] >> 8))
                p0 = fbase[t0[0] + n] * fbase[t0[1] + n];
            if (n >= (t1[2] & 0xFF) && n < (t1[2] >> 8))
                p1 = fbase[t1[0] + n] * fbase[t1[1] + n];
            *(unsigned int*)(Bs + n*32 + 2*klp) = f2bf2(p0, p1);
        }
        cp_drain();
        __syncthreads();

        v16bf afrag = load_frag_a(As, strip*16 + li, g);
#pragma unroll
        for (int nt = 0; nt < 3; ++nt) {
            v16bf bfrag = load_frag_b(Bs, nh*48 + nt*16 + li, g);
            acc[nt] = __builtin_amdgcn_wmma_f32_16x16x32_bf16(
                false, afrag, false, bfrag, (short)0, acc[nt], false, false);
        }
        __syncthreads();
    }

#pragma unroll
    for (int nt = 0; nt < 3; ++nt) {
#pragma unroll
        for (int r = 0; r < 8; ++r) {
            int m = strip*16 + r + 8*g;
            int n = nh*48 + nt*16 + li;
            y[(((size_t)b*OC_ + m)*H_ + i)*W_ + n] = acc[nt][r] + bm[r];
        }
    }
}

extern "C" void kernel_launch(void* const* d_in, const int* in_sizes, int n_in,
                              void* d_out, int out_size, void* d_ws, size_t ws_size,
                              hipStream_t stream) {
    const float* x  = (const float*)d_in[0];   // (4,256,96,96)
    const float* ew = (const float*)d_in[1];   // (64,256,3,3)
    const float* eb = (const float*)d_in[2];   // (64,)
    const float* rw = (const float*)d_in[3];   // (32,64,5,5)
    const float* rb = (const float*)d_in[4];   // (32,)
    float* y = (float*)d_out;                  // (4,32,96,96)

    float*          f   = (float*)((char*)d_ws + WS_F_OFF);    // (4,64,96,96) f32
    unsigned short* ewb = (unsigned short*)((char*)d_ws + WS_EWB_OFF);
    unsigned short* rwb = (unsigned short*)((char*)d_ws + WS_RWB_OFF);

    pack_weights<<<(EW_PAIRS + RW_PAIRS + 255)/256, 256, 0, stream>>>(ew, rw, ewb, rwb);
    dim3 grid(B_ * H_), blk(128);
    conv3x3_wmma<<<grid, blk, 0, stream>>>(x, ewb, eb, f);
    corrproj_wmma<<<grid, blk, 0, stream>>>(f, rwb, rb, y);
}
